// DeepSeekMoE_26199300505737
// MI455X (gfx1250) — compile-verified, matrix operand order checked by reading, not run on hardware
//
#include <hip/hip_runtime.h>
#include <hip/hip_bf16.h>

// dims from reference
#define T_TOK 2048   // B*S
#define DM    1024   // d_model
#define HD    512    // d_hidden
#define ER    64     // routed experts
#define TK    6      // top-k
#define ES    2      // shared experts
#define EPSF  1.1920929e-07f
#define BSTR  40     // padded k-stride (elements) of transposed weight tile: 80B rows, 16B aligned

#define AS1 __attribute__((address_space(1)))
#define AS3 __attribute__((address_space(3)))

#if __has_builtin(__builtin_amdgcn_global_load_async_to_lds_b128)
#define HAVE_ASYNC 1
#endif

typedef int vsi4 __attribute__((vector_size(16)));   // matches builtin param pointee type
typedef __attribute__((ext_vector_type(16))) __bf16         v16bf;
typedef __attribute__((ext_vector_type(16))) unsigned short v16u;
typedef __attribute__((ext_vector_type(8)))  unsigned short v8u;
typedef __attribute__((ext_vector_type(8)))  float          v8f;

__device__ __forceinline__ unsigned short f2bf(float f) {
    unsigned int u = __builtin_bit_cast(unsigned int, f);
    return (unsigned short)((u + 0x8000u) >> 16);          // round-to-nearest
}
__device__ __forceinline__ float bf2f(unsigned short s) {
    unsigned int u = ((unsigned int)s) << 16;
    return __builtin_bit_cast(float, u);
}
// pack two f32 -> two bf16 in one dword: 2x v_add + 1x v_perm
#if __has_builtin(__builtin_amdgcn_cvt_pk_bf16_f32)
typedef __attribute__((ext_vector_type(2))) __bf16 v2bf;
__device__ __forceinline__ unsigned int pk2bf(float a, float b) {
    v2bf v = __builtin_amdgcn_cvt_pk_bf16_f32(a, b);
    return __builtin_bit_cast(unsigned int, v);
}
#else
__device__ __forceinline__ unsigned int pk2bf(float a, float b) {
    unsigned int ua = __builtin_bit_cast(unsigned int, a) + 0x8000u;
    unsigned int ub = __builtin_bit_cast(unsigned int, b) + 0x8000u;
    return __builtin_amdgcn_perm(ub, ua, 0x07060302u);     // {ub.hi16, ua.hi16}
}
#endif
__device__ __forceinline__ v8f vzero8() {
    v8f v;
    #pragma unroll
    for (int i = 0; i < 8; ++i) v[i] = 0.0f;
    return v;
}

#ifdef HAVE_ASYNC
__device__ __forceinline__ void async_cp16(const unsigned short* g,
                                           unsigned short* l) {
    __builtin_amdgcn_global_load_async_to_lds_b128(
        (AS1 vsi4*)(size_t)g,
        (AS3 vsi4*)(unsigned int)(size_t)l,
        0, 0);
}
__device__ __forceinline__ void wait_async0() {
#if __has_builtin(__builtin_amdgcn_s_wait_asynccnt)
    __builtin_amdgcn_s_wait_asynccnt(0);
#else
    asm volatile("s_wait_asynccnt 0x0" ::: "memory");
#endif
}
#endif

// stage one 16-byte A segment: async DMA to LDS if available, else sync copy
__device__ __forceinline__ void stage_a16(const unsigned short* g, unsigned short* l) {
#ifdef HAVE_ASYNC
    async_cp16(g, l);
#else
    *(v8u*)l = *(const v8u*)g;
#endif
}
__device__ __forceinline__ void stage_a_fence() {
#ifdef HAVE_ASYNC
    wait_async0();
#endif
}

// Load a 16x32 bf16 fragment (A: major=M, minor=K | B: major=N, minor=K).
// ISA 16-bit layout: lanes 0-15 hold K {0..7,16..23}, lanes 16-31 hold K {8..15,24..31}.
__device__ __forceinline__ v16bf ld_frag(const unsigned short* __restrict__ base,
                                         int major0, int ld, int minor0, int lane) {
    int major = major0 + (lane & 15);
    int off   = (lane >> 4) << 3;                  // 0 or 8
    const unsigned short* p = base + major * ld + minor0 + off;
    v8u lo = *(const v8u*)p;                       // ds_load_b128
    v8u hi = *(const v8u*)(p + 16);                // ds_load_b128
    v16u r;
    #pragma unroll
    for (int i = 0; i < 8; ++i) { r[i] = lo[i]; r[i + 8] = hi[i]; }
    return __builtin_bit_cast(v16bf, r);
}

__device__ __forceinline__ v8f wmma_bf16(v16bf a, v16bf b, v8f c) {
    return __builtin_amdgcn_wmma_f32_16x16x32_bf16(false, a, false, b,
                                                   (short)0, c, false, false);
}

// ---------------- kernel 1: out = x (residual); zero expert counters ----------------
__global__ void k_init(const float* __restrict__ x, float* __restrict__ out, int n,
                       int* __restrict__ counts) {
    int i = blockIdx.x * blockDim.x + threadIdx.x;
    if (i < n)  out[i] = x[i];
    if (i < ER) counts[i] = 0;
}

// ---------------- kernel 2: RMSNorm -> bf16 tokens (vectorized) ----------------
__global__ __launch_bounds__(256) void k_rmsnorm(const float* __restrict__ x,
                                                 const float* __restrict__ nw,
                                                 unsigned short* __restrict__ xn) {
    int t = blockIdx.x;
    int tid = threadIdx.x;
    __shared__ float red[256];
    int d0 = tid * 4;
    float4 v = *(const float4*)(x + (size_t)t * DM + d0);
    float4 w = *(const float4*)(nw + d0);
    float ss = v.x * v.x + v.y * v.y + v.z * v.z + v.w * v.w;
    red[tid] = ss;
    __syncthreads();
    for (int s = 128; s > 0; s >>= 1) {
        if (tid < s) red[tid] += red[tid + s];
        __syncthreads();
    }
    float r = rsqrtf(red[0] * (1.0f / DM) + EPSF);
    uint2 o;
    o.x = pk2bf(v.x * r * w.x, v.y * r * w.y);
    o.y = pk2bf(v.z * r * w.z, v.w * r * w.w);
    *(uint2*)(xn + (size_t)t * DM + d0) = o;
}

// ---------------- kernel 3: router via WMMA + per-token top-k ----------------
// grid = T/32 tiles; block = 128 = 4 waves; wave w owns expert-cols [w*16, w*16+16)
__global__ __launch_bounds__(128) void k_router(const unsigned short* __restrict__ xn,
                                                const float* __restrict__ Wr,
                                                const float* __restrict__ bias,
                                                int* __restrict__ topk_idx,
                                                float* __restrict__ topk_w,
                                                int* __restrict__ counts) {
    int tile = blockIdx.x;
    int tid  = threadIdx.x;
    int lane = tid & 31;
    int wave = tid >> 5;          // 0..3
    int n0   = wave * 16;

    __shared__ unsigned short aT[32 * 32];       // token tile (M x Kchunk)
    __shared__ unsigned short bW[ER * BSTR];     // Wr^T tile  (N x Kchunk, padded)
    __shared__ float rawS[32 * ER];              // raw affinities

    int arow = tid >> 2, aseg = tid & 3;         // 128 threads -> 32 rows x 4 segs
    const unsigned short* pA = xn + (size_t)(tile * 32 + arow) * DM + aseg * 8;
    unsigned short* lA = aT + arow * 32 + aseg * 8;

    v8f acc[2];
    acc[0] = vzero8(); acc[1] = vzero8();

    for (int k0 = 0; k0 < DM; k0 += 32) {
        stage_a16(pA + k0, lA);                         // async -> LDS (bf16 copy)
        // issue all 4 weight loads first (distinct temps), then pack+store
        float4 t[4];
        #pragma unroll
        for (int j = 0; j < 4; ++j) {
            int i = tid + j * 128;
            int n = i >> 3, ks = (i & 7) * 4;
            t[j] = *(const float4*)(Wr + (size_t)n * DM + k0 + ks); // rows of Wr: contiguous
        }
        #pragma unroll
        for (int j = 0; j < 4; ++j) {
            int i = tid + j * 128;
            int n = i >> 3, ks = (i & 7) * 4;
            uint2 u;
            u.x = pk2bf(t[j].x, t[j].y);
            u.y = pk2bf(t[j].z, t[j].w);
            *(uint2*)(bW + n * BSTR + ks) = u;
        }
        stage_a_fence();
        __syncthreads();
        v16bf bfr = ld_frag(bW, n0, BSTR, 0, lane);
        acc[0] = wmma_bf16(ld_frag(aT, 0,  32, 0, lane), bfr, acc[0]);
        acc[1] = wmma_bf16(ld_frag(aT, 16, 32, 0, lane), bfr, acc[1]);
        __syncthreads();
    }
    int mro = (lane >> 4) << 3, ncl = lane & 15;
    #pragma unroll
    for (int mt = 0; mt < 2; ++mt)
        #pragma unroll
        for (int r = 0; r < 8; ++r)
            rawS[(mt * 16 + mro + r) * ER + n0 + ncl] = acc[mt][r];
    __syncthreads();

    if (tid < 32) {
        int t = tile * 32 + tid;
        const float* rw = rawS + tid * ER;
        float aff[ER];
        for (int i = 0; i < ER; ++i) aff[i] = rw[i] + bias[i];
        int   idx[TK];
        float rk[TK];
        float s = 0.0f;
        for (int k = 0; k < TK; ++k) {
            int bi = 0; float bv = -3.4e38f;
            for (int i = 0; i < ER; ++i)
                if (aff[i] > bv) { bv = aff[i]; bi = i; }
            idx[k] = bi; rk[k] = rw[bi]; aff[bi] = -3.4e38f; s += rk[k];
        }
        float inv = 1.0f / s;
        for (int k = 0; k < TK; ++k) {
            topk_idx[t * TK + k] = idx[k];
            topk_w[t * TK + k]   = rk[k] * inv;
            atomicAdd(&counts[idx[k]], 1);
        }
    }
}

// ---------------- kernel 4: exclusive scan of counts ----------------
__global__ void k_scan(const int* __restrict__ counts, int* __restrict__ offsets,
                       int* __restrict__ cursor) {
    if (threadIdx.x == 0) {
        int acc = 0;
        for (int e = 0; e < ER; ++e) {
            offsets[e] = acc;
            cursor[e]  = acc;
            acc += counts[e];
        }
    }
}

// ---------------- kernel 5: scatter tokens into expert buckets ----------------
__global__ void k_scatter(const int* __restrict__ topk_idx,
                          const float* __restrict__ topk_w,
                          int* __restrict__ cursor,
                          int* __restrict__ btok, float* __restrict__ bw) {
    int t = blockIdx.x * blockDim.x + threadIdx.x;
    if (t >= T_TOK) return;
    for (int k = 0; k < TK; ++k) {
        int e = topk_idx[t * TK + k];
        int pos = atomicAdd(&cursor[e], 1);
        btok[pos] = t;
        bw[pos]   = topk_w[t * TK + k];
    }
}

// stage a 32xHD weight chunk (fp32, row stride ldW) transposed+bf16 into bT; 256 threads.
// software-pipelined: per group, issue 16 loads into distinct temps, then pack+store.
__device__ __forceinline__ void stage_w(const float* __restrict__ W, int ldW,
                                        unsigned short* __restrict__ bT, int tid) {
    #pragma unroll
    for (int g = 0; g < 4; ++g) {
        float t[4][4];
        #pragma unroll
        for (int j = 0; j < 4; ++j) {
            int i = tid + (g * 4 + j) * 256;
            int n = i & (HD - 1), kk = (i >> 9) << 2;
            const float* wp = W + (size_t)kk * ldW + n;  // coalesced over n
            t[j][0] = wp[0];
            t[j][1] = wp[ldW];
            t[j][2] = wp[2 * ldW];
            t[j][3] = wp[3 * ldW];
        }
        #pragma unroll
        for (int j = 0; j < 4; ++j) {
            int i = tid + (g * 4 + j) * 256;
            int n = i & (HD - 1), kk = (i >> 9) << 2;
            uint2 u;
            u.x = pk2bf(t[j][0], t[j][1]);
            u.y = pk2bf(t[j][2], t[j][3]);
            *(uint2*)(bT + n * BSTR + kk) = u;
        }
    }
}

// ---------------- kernel 6: expert FFN tile (SwiGLU), WMMA bf16, H in registers ----------------
// block = 256 threads = 8 waves; M-tile = 32 tokens; wave owns 64 N-columns.
// phase1: H = X(32xDM) @ W1 + b1      (H kept in hreg, bf16 copy -> hB)
// phase2: G = H @ Wg ; H' = silu(G)*H (register gate, overwrite hB)
// phase3: O = H' @ W2 + b2 ; out[tok] += w_tok * O  (atomic f32)
__global__ __launch_bounds__(256) void k_ffn(
    const unsigned short* __restrict__ xn,
    const float* __restrict__ W1, const float* __restrict__ b1,
    const float* __restrict__ Wg,
    const float* __restrict__ W2, const float* __restrict__ b2,
    const int* __restrict__ toklist, const float* __restrict__ wlist,
    const int* __restrict__ counts, const int* __restrict__ offsets,
    float* __restrict__ out, int routed) {

    const int e    = blockIdx.y;
    const int tile = blockIdx.x;
    const int tid  = threadIdx.x;
    const int lane = tid & 31;
    const int wave = tid >> 5;          // 0..7
    const int n0   = wave * 64;

    int cnt = 32;
    const int*   tl = nullptr;
    const float* wl = nullptr;
    if (routed) {
        int c = counts[e];
        int ntiles = (c + 31) >> 5;
        if (tile >= ntiles) return;
        int base = offsets[e] + tile * 32;
        tl = toklist + base;
        wl = wlist   + base;
        cnt = c - tile * 32;
        if (cnt > 32) cnt = 32;
    }

    const float* W1e = W1 + (size_t)e * DM * HD;
    const float* b1e = b1 + (size_t)e * HD;
    const float* W2e = W2 + (size_t)e * HD * DM;
    const float* b2e = b2 + (size_t)e * DM;

    extern __shared__ char smem[];
    unsigned short* aT = (unsigned short*)smem;                  // 32 x 32   bf16 (M x Kchunk)
    unsigned short* bT = (unsigned short*)(smem + 2048);         // 512 x 40  bf16 (N x Kchunk, padded)
    unsigned short* hB = (unsigned short*)(smem + 2048 + HD * BSTR * 2); // 32 x 512 bf16

    __shared__ int   rowTok[32];
    __shared__ float rowW[32];
    if (tid < 32) {
        if (routed) {
            if (tid < cnt) { rowTok[tid] = tl[tid];  rowW[tid] = wl[tid]; }
            else           { rowTok[tid] = tl[0];    rowW[tid] = 0.0f;    }
        } else {
            rowTok[tid] = tile * 32 + tid;
            rowW[tid]   = 1.0f;
        }
    }
    __syncthreads();

    // hoisted A-staging pointers: threads 0..127 -> (row, 8-elem segment)
    const int arow = tid >> 2, aseg = tid & 3;
    const unsigned short* pA =
        (tid < 128) ? (xn + (size_t)rowTok[arow] * DM + aseg * 8) : nullptr;
    unsigned short* lA = aT + arow * 32 + aseg * 8;

    const int mro = (lane >> 4) << 3;   // C-layout row offset: 0 or 8
    const int ncl = lane & 15;          // C-layout column within 16

    v8f acc[2][4], hreg[2][4];
    #pragma unroll
    for (int mt = 0; mt < 2; ++mt)
        #pragma unroll
        for (int nt = 0; nt < 4; ++nt) acc[mt][nt] = vzero8();

    // ---- phase 1: X @ W1 ----
    for (int k0 = 0; k0 < DM; k0 += 32) {
        if (tid < 128) stage_a16(pA + k0, lA);           // async bf16 token copy
        stage_w(W1e + (size_t)k0 * HD, HD, bT, tid);
        stage_a_fence();
        __syncthreads();
        v16bf bfr[4];
        #pragma unroll
        for (int nt = 0; nt < 4; ++nt) bfr[nt] = ld_frag(bT, n0 + nt * 16, BSTR, 0, lane);
        #pragma unroll
        for (int mt = 0; mt < 2; ++mt) {
            v16bf afr = ld_frag(aT, mt * 16, 32, 0, lane);
            #pragma unroll
            for (int nt = 0; nt < 4; ++nt)
                acc[mt][nt] = wmma_bf16(afr, bfr[nt], acc[mt][nt]);
        }
        __syncthreads();
    }
    // H -> registers (+b1), bf16 copy -> hB
    #pragma unroll
    for (int mt = 0; mt < 2; ++mt)
        #pragma unroll
        for (int nt = 0; nt < 4; ++nt) {
            int n = n0 + nt * 16 + ncl;
            float bv = b1e[n];
            #pragma unroll
            for (int r = 0; r < 8; ++r) {
                int m = mt * 16 + mro + r;
                float h = acc[mt][nt][r] + bv;
                hreg[mt][nt][r] = h;
                hB[m * HD + n] = f2bf(h);
            }
            acc[mt][nt] = vzero8();
        }
    __syncthreads();

    // ---- phase 2: G = H @ Wg ; register gate ----
    for (int k0 = 0; k0 < HD; k0 += 32) {
        stage_w(Wg + (size_t)k0 * HD, HD, bT, tid);
        __syncthreads();
        v16bf bfr[4];
        #pragma unroll
        for (int nt = 0; nt < 4; ++nt) bfr[nt] = ld_frag(bT, n0 + nt * 16, BSTR, 0, lane);
        #pragma unroll
        for (int mt = 0; mt < 2; ++mt) {
            v16bf afr = ld_frag(hB, mt * 16, HD, k0, lane);
            #pragma unroll
            for (int nt = 0; nt < 4; ++nt)
                acc[mt][nt] = wmma_bf16(afr, bfr[nt], acc[mt][nt]);
        }
        __syncthreads();
    }
    // gate in registers, overwrite hB (reads finished at last barrier)
    #pragma unroll
    for (int mt = 0; mt < 2; ++mt)
        #pragma unroll
        for (int nt = 0; nt < 4; ++nt) {
            int n = n0 + nt * 16 + ncl;
            #pragma unroll
            for (int r = 0; r < 8; ++r) {
                int m = mt * 16 + mro + r;
                float g = acc[mt][nt][r];
                float sg = g / (1.0f + __expf(-g));
                hB[m * HD + n] = f2bf(sg * hreg[mt][nt][r]);
            }
        }
    __syncthreads();

    // ---- phase 3: O = H' @ W2 (+b2), scaled accumulate ----
    for (int nh = 0; nh < 2; ++nh) {
        #pragma unroll
        for (int mt = 0; mt < 2; ++mt)
            #pragma unroll
            for (int nt = 0; nt < 4; ++nt) acc[mt][nt] = vzero8();
        for (int k0 = 0; k0 < HD; k0 += 32) {
            stage_w(W2e + (size_t)k0 * DM + nh * HD, DM, bT, tid);
            __syncthreads();
            v16bf bfr[4];
            #pragma unroll
            for (int nt = 0; nt < 4; ++nt) bfr[nt] = ld_frag(bT, n0 + nt * 16, BSTR, 0, lane);
            #pragma unroll
            for (int mt = 0; mt < 2; ++mt) {
                v16bf afr = ld_frag(hB, mt * 16, HD, k0, lane);
                #pragma unroll
                for (int nt = 0; nt < 4; ++nt)
                    acc[mt][nt] = wmma_bf16(afr, bfr[nt], acc[mt][nt]);
            }
            __syncthreads();
        }
        #pragma unroll
        for (int mt = 0; mt < 2; ++mt)
            #pragma unroll
            for (int nt = 0; nt < 4; ++nt)
                #pragma unroll
                for (int r = 0; r < 8; ++r) {
                    int m = mt * 16 + mro + r;
                    if (routed && m >= cnt) continue;
                    int col = nh * HD + n0 + nt * 16 + ncl;
                    float val = (acc[mt][nt][r] + b2e[col]) * rowW[m];
                    atomicAdd(&out[(size_t)rowTok[m] * DM + col], val);
                }
    }
}

// ---------------- host launch ----------------
extern "C" void kernel_launch(void* const* d_in, const int* in_sizes, int n_in,
                              void* d_out, int out_size, void* d_ws, size_t ws_size,
                              hipStream_t stream) {
    (void)in_sizes; (void)n_in; (void)out_size; (void)ws_size;
    const float* x      = (const float*)d_in[0];
    const float* norm_w = (const float*)d_in[1];
    const float* Wr     = (const float*)d_in[2];
    const float* sW1    = (const float*)d_in[3];
    const float* sb1    = (const float*)d_in[4];
    const float* sW2    = (const float*)d_in[5];
    const float* sb2    = (const float*)d_in[6];
    const float* sWg    = (const float*)d_in[7];
    const float* rW1    = (const float*)d_in[8];
    const float* rb1    = (const float*)d_in[9];
    const float* rW2    = (const float*)d_in[10];
    const float* rb2    = (const float*)d_in[11];
    const float* rWg    = (const float*)d_in[12];
    const float* bias   = (const float*)d_in[13];
    float* out = (float*)d_out;

    char* ws = (char*)d_ws;
    unsigned short* xn = (unsigned short*)ws;  ws += (size_t)T_TOK * DM * 2;
    int*   topk_idx = (int*)ws;                ws += (size_t)T_TOK * TK * 4;
    float* topk_w   = (float*)ws;              ws += (size_t)T_TOK * TK * 4;
    int*   counts   = (int*)ws;                ws += ER * 4;
    int*   offsets  = (int*)ws;                ws += ER * 4;
    int*   cursor   = (int*)ws;                ws += ER * 4;
    int*   btok     = (int*)ws;                ws += (size_t)T_TOK * TK * 4;
    float* bw       = (float*)ws;              ws += (size_t)T_TOK * TK * 4;

    k_init<<<(T_TOK * DM + 255) / 256, 256, 0, stream>>>(x, out, T_TOK * DM, counts);
    k_rmsnorm<<<T_TOK, 256, 0, stream>>>(x, norm_w, xn);
    k_router<<<T_TOK / 32, 128, 0, stream>>>(xn, Wr, bias, topk_idx, topk_w, counts);
    k_scan<<<1, 64, 0, stream>>>(counts, offsets, cursor);
    k_scatter<<<(T_TOK + 255) / 256, 256, 0, stream>>>(topk_idx, topk_w, cursor, btok, bw);

    size_t smem = 2048 + (size_t)HD * BSTR * 2 + (size_t)32 * HD * 2;  // aT + bT + hB = 74 KB
    // shared experts: all tokens, weight 1.0
    k_ffn<<<dim3(T_TOK / 32, ES), 256, smem, stream>>>(
        xn, sW1, sb1, sWg, sW2, sb2,
        nullptr, nullptr, nullptr, nullptr, out, 0);
    // routed experts: bucketed tokens, routing weights
    k_ffn<<<dim3(T_TOK / 32, ER), 256, smem, stream>>>(
        xn, rW1, rb1, rWg, rW2, rb2,
        btok, bw, counts, offsets, out, 1);
}
